// StockLSTM_78907139162366
// MI455X (gfx1250) — matrix operand order repaired and checked by reading, not compiled, not run on hardware
//
#include <hip/hip_runtime.h>
#include <hip/hip_bf16.h>

typedef _Float16 h16;
typedef h16   v16h __attribute__((ext_vector_type(16)));
typedef h16   v8h  __attribute__((ext_vector_type(8)));
typedef float v8f  __attribute__((ext_vector_type(8)));

#define T_   512
#define D_   5
#define H_   64
#define G_   256
#define FC1_ 32

#if __has_builtin(__builtin_amdgcn_tanhf)
__device__ __forceinline__ float tanh_(float v) { return __builtin_amdgcn_tanhf(v); }
#else
__device__ __forceinline__ float tanh_(float v) {
    return 1.0f - 2.0f * __builtin_amdgcn_rcpf(__expf(2.0f * v) + 1.0f);
}
#endif
// sigmoid via hardware tanh: sig(x) = 0.5 + 0.5*tanh(x/2)
__device__ __forceinline__ float sigm_(float v) { return fmaf(0.5f, tanh_(0.5f * v), 0.5f); }

__device__ __forceinline__ v8f wmma_f16(v16h a, v16h b, v8f c) {
    return __builtin_amdgcn_wmma_f32_16x16x32_f16(false, a, false, b, (short)0, c, false, false);
}

// A-matrix 16x32 f16 fragment from a row-major [16][ldk] f16 LDS tile.
// Lane l: row (l&15); lanes 0-15 K 0..15, lanes 16-31 K 16..31.
__device__ __forceinline__ v16h load_a(const h16* tile, int ldk, int kf) {
    int lane = threadIdx.x & 31;
    int row  = lane & 15;
    int koff = (lane < 16) ? 0 : 16;
    const h16* p = tile + row * ldk + kf * 32 + koff;
    v8h lo = *(const v8h*)(p);
    v8h hi = *(const v8h*)(p + 8);
    return __builtin_shufflevector(lo, hi, 0,1,2,3,4,5,6,7,8,9,10,11,12,13,14,15);
}

// B-matrix 32x16 f16 fragment converted from row-major global fp32 W[g][ldk] (g = N index).
// Lane l: column ntile*16+(l&15); lanes 0-15 K 0..15, lanes 16-31 K 16..31 of the K-slab.
__device__ __forceinline__ v16h load_b_g(const float* W, int ldk, int kf, int ntile) {
    int lane  = threadIdx.x & 31;
    int n     = ntile * 16 + (lane & 15);
    int kbase = kf * 32 + ((lane < 16) ? 0 : 16);
    const float* p = W + n * ldk + kbase;
    v16h r;
#pragma unroll
    for (int i = 0; i < 16; ++i) r[i] = (h16)p[i];
    return r;
}

// W_ih0 fragment: K padded 5->32, with combined layer-0 bias folded into row k==5
// (the staged x tile carries a constant 1.0 in column 5).
__device__ __forceinline__ v16h load_b_wih0(const float* Wih0, const float* bih0,
                                            const float* bhh0, int ntile) {
    int lane  = threadIdx.x & 31;
    int n     = ntile * 16 + (lane & 15);
    int kbase = (lane < 16) ? 0 : 16;
    v16h r;
#pragma unroll
    for (int i = 0; i < 16; ++i) {
        int k = kbase + i;
        float v = 0.0f;
        if (k < D_)       v = Wih0[n * D_ + k];
        else if (k == D_) v = bih0[n] + bhh0[n];
        r[i] = (h16)v;
    }
    return r;
}

__global__ __launch_bounds__(128, 1)
void lstm2_fused_kernel(const float* __restrict__ x,
                        const float* __restrict__ Wih0, const float* __restrict__ Whh0,
                        const float* __restrict__ bih0, const float* __restrict__ bhh0,
                        const float* __restrict__ Wih1, const float* __restrict__ Whh1,
                        const float* __restrict__ bih1, const float* __restrict__ bhh1,
                        const float* __restrict__ W1,   const float* __restrict__ b1,
                        const float* __restrict__ W2,   const float* __restrict__ b2,
                        float* __restrict__ out) {
    __shared__ __align__(16) h16 s_h1[2][16 * H_];   // ping-pong h1 tiles, f16 row-major
    __shared__ __align__(16) h16 s_h2[2][16 * H_];   // ping-pong h2 tiles
    __shared__ __align__(16) h16 s_x [16 * 32];      // x_t tile, K padded to 32, col 5 == 1.0

    const int tid  = threadIdx.x;
    const int wave = tid >> 5;
    const int lane = tid & 31;
    const int rowBase = blockIdx.x * 16;
    const int jcol  = wave * 16 + (lane & 15);   // hidden column this lane owns
    const int mbase = (lane < 16) ? 0 : 8;       // C-layout row base per half-wave

    // ---- zero/seed LDS tiles ----
    for (int i = tid; i < 16 * H_; i += 128) {
        s_h1[0][i] = (h16)0.0f; s_h1[1][i] = (h16)0.0f;
        s_h2[0][i] = (h16)0.0f; s_h2[1][i] = (h16)0.0f;
    }
    for (int i = tid; i < 16 * 32; i += 128)
        s_x[i] = ((i & 31) == D_) ? (h16)1.0f : (h16)0.0f;   // constant-1 bias column

    // ---- loop-invariant weight fragments in registers (f16, converted from global fp32) ----
    v16h wih0f[4];        // layer-0 input weights + folded bias, K padded to 32
    v16h whh0f[4][2];     // layer-0 recurrent weights, K = 64 -> 2 frags
    v16h wih1f[4][2];     // layer-1 input weights
    v16h whh1f[4][2];     // layer-1 recurrent weights
    v8f  bias1v[4];       // layer-1 combined bias, splat per gate tile
#pragma unroll
    for (int gi = 0; gi < 4; ++gi) {
        int nt = gi * 4 + wave;
        wih0f[gi]    = load_b_wih0(Wih0, bih0, bhh0, nt);
        whh0f[gi][0] = load_b_g(Whh0, H_, 0, nt);
        whh0f[gi][1] = load_b_g(Whh0, H_, 1, nt);
        wih1f[gi][0] = load_b_g(Wih1, H_, 0, nt);
        wih1f[gi][1] = load_b_g(Wih1, H_, 1, nt);
        whh1f[gi][0] = load_b_g(Whh1, H_, 0, nt);
        whh1f[gi][1] = load_b_g(Whh1, H_, 1, nt);
        int idx = gi * H_ + jcol;
        float bb = bih1[idx] + bhh1[idx];
        bias1v[gi] = (v8f){bb, bb, bb, bb, bb, bb, bb, bb};
    }

    v8f c1 = {};   // layer-0 cell state tile (16 rows x this wave's 16 cols)
    v8f c2 = {};   // layer-1 cell state tile
    __syncthreads();

    for (int t = 0; t < T_; ++t) {
        const int pg = t & 1, pr = pg ^ 1;

        // ---- stage x_t (16 rows x 5 features) into padded f16 tile ----
        if (tid < 16 * D_) {
            int r = tid / D_, d = tid - r * D_;
            s_x[r * 32 + d] = (h16)x[(size_t)(rowBase + r) * (T_ * D_) + t * D_ + d];
        }
        __syncthreads();   // B1: x staged; h1[t-1]/h2[t-1] stores visible

        // ========= layer 0: gates = [x_t|1] @ [Wih0|b0]^T + h1[t-1] @ Whh0^T =========
        v8f acc[4];
        {
            v16h xa  = load_a(s_x,       32, 0);
            v16h ha0 = load_a(s_h1[pr],  H_, 0);
            v16h ha1 = load_a(s_h1[pr],  H_, 1);
#pragma unroll
            for (int gi = 0; gi < 4; ++gi) {
                v8f zc = {};
                v8f a = wmma_f16(xa,  wih0f[gi],    zc);
                a     = wmma_f16(ha0, whh0f[gi][0], a);
                a     = wmma_f16(ha1, whh0f[gi][1], a);
                acc[gi] = a;
            }
        }
        v8f hnew;
#pragma unroll
        for (int r = 0; r < 8; ++r) {
            float iv = sigm_(acc[0][r]);
            float fv = sigm_(acc[1][r]);
            float gv = tanh_(acc[2][r]);
            float ov = sigm_(acc[3][r]);
            float c  = fv * c1[r] + iv * gv;
            c1[r] = c;
            hnew[r] = ov * tanh_(c);
        }
#pragma unroll
        for (int r = 0; r < 8; ++r)       // write tile != read tile: no WAR barrier needed
            s_h1[pg][(mbase + r) * H_ + jcol] = (h16)hnew[r];
        __syncthreads();   // B2: h1[t] visible

        // ========= layer 1: gates = b1 + h1[t] @ Wih1^T + h2[t-1] @ Whh1^T =========
        {
            v16h hb0 = load_a(s_h1[pg], H_, 0);
            v16h hb1 = load_a(s_h1[pg], H_, 1);
            v16h g0  = load_a(s_h2[pr], H_, 0);
            v16h g1  = load_a(s_h2[pr], H_, 1);
#pragma unroll
            for (int gi = 0; gi < 4; ++gi) {
                v8f a = wmma_f16(hb0, wih1f[gi][0], bias1v[gi]);
                a     = wmma_f16(hb1, wih1f[gi][1], a);
                a     = wmma_f16(g0,  whh1f[gi][0], a);
                a     = wmma_f16(g1,  whh1f[gi][1], a);
                acc[gi] = a;
            }
        }
#pragma unroll
        for (int r = 0; r < 8; ++r) {
            float iv = sigm_(acc[0][r]);
            float fv = sigm_(acc[1][r]);
            float gv = tanh_(acc[2][r]);
            float ov = sigm_(acc[3][r]);
            float c  = fv * c2[r] + iv * gv;
            c2[r] = c;
            hnew[r] = ov * tanh_(c);
        }
#pragma unroll
        for (int r = 0; r < 8; ++r)
            s_h2[pg][(mbase + r) * H_ + jcol] = (h16)hnew[r];
        // loop back: next B1 covers h-store visibility before anyone reads
    }
    __syncthreads();   // final h2[T-1] (buffer (T_-1)&1 == 1) visible

    // ---- FC head: out = relu(h2_T @ W1^T + b1) @ W2^T + b2, one row per thread ----
    if (tid < 16) {
        const h16* hT = s_h2[(T_ - 1) & 1];
        float o = b2[0];
        for (int k2 = 0; k2 < FC1_; ++k2) {
            float z = b1[k2];
            for (int k = 0; k < H_; ++k)
                z += W1[k2 * H_ + k] * (float)hT[tid * H_ + k];
            o += W2[k2] * fmaxf(z, 0.0f);
        }
        out[rowBase + tid] = o;
    }
}

extern "C" void kernel_launch(void* const* d_in, const int* in_sizes, int n_in,
                              void* d_out, int out_size, void* d_ws, size_t ws_size,
                              hipStream_t stream) {
    (void)in_sizes; (void)n_in; (void)d_ws; (void)ws_size; (void)out_size;
    const float* x    = (const float*)d_in[0];
    const float* Wih0 = (const float*)d_in[1];
    const float* Whh0 = (const float*)d_in[2];
    const float* bih0 = (const float*)d_in[3];
    const float* bhh0 = (const float*)d_in[4];
    const float* Wih1 = (const float*)d_in[5];
    const float* Whh1 = (const float*)d_in[6];
    const float* bih1 = (const float*)d_in[7];
    const float* bhh1 = (const float*)d_in[8];
    const float* W1   = (const float*)d_in[9];
    const float* b1   = (const float*)d_in[10];
    const float* W2   = (const float*)d_in[11];
    const float* b2   = (const float*)d_in[12];
    float* out = (float*)d_out;

    dim3 grid(4096 / 16);   // 256 batch tiles
    dim3 block(128);        // 4 wave32s
    lstm2_fused_kernel<<<grid, block, 0, stream>>>(x, Wih0, Whh0, bih0, bhh0,
                                                   Wih1, Whh1, bih1, bhh1,
                                                   W1, b1, W2, b2, out);
}
